// VisionSelfAttention_27728308863766
// MI455X (gfx1250) — matrix operand order repaired and checked by reading, not compiled, not run on hardware
//
#include <hip/hip_runtime.h>
#include <hip/hip_bf16.h>

typedef __attribute__((ext_vector_type(16))) __bf16 v16bf;
typedef __attribute__((ext_vector_type(8)))  float  v8f;
typedef __attribute__((ext_vector_type(4)))  int    i32x4;

#define PP   8
#define NN   1024
#define DD   768
#define HH   12
#define HS   64
#define SCALE 8.0f   // sqrt(64)

// ---------------------------------------------------------------- async global->LDS
#if defined(__has_builtin)
#if __has_builtin(__builtin_amdgcn_global_load_async_to_lds_b128)
#define USE_ASYNC 1
#endif
#endif
#ifndef USE_ASYNC
#define USE_ASYNC 0
#endif

__device__ __forceinline__ void async_b128(void* lds, const void* g) {
#if USE_ASYNC
    __builtin_amdgcn_global_load_async_to_lds_b128(
        (__attribute__((address_space(1))) i32x4*)g,
        (__attribute__((address_space(3))) i32x4*)lds, 0, 0);
#else
    *(uint4*)lds = *(const uint4*)g;
#endif
}

__device__ __forceinline__ void async_wait0() {
#if USE_ASYNC
#if defined(__has_builtin) && __has_builtin(__builtin_amdgcn_s_wait_asynccnt)
    __builtin_amdgcn_s_wait_asynccnt(0);
#else
    asm volatile("s_wait_asynccnt 0x0" ::: "memory");
#endif
#else
    asm volatile("" ::: "memory");
#endif
}

// ---------------------------------------------------------------- cvt f32->bf16
__global__ __launch_bounds__(256) void cvt_kernel(const float* __restrict__ in,
                                                  __bf16* __restrict__ out, int n) {
    int i = blockIdx.x * blockDim.x + threadIdx.x;
    int stride = gridDim.x * blockDim.x;
    for (; i < n; i += stride) out[i] = (__bf16)in[i];
}

// cvt + transpose: in (Kd x Nd) row-major f32 -> out (Nd x Kd) row-major bf16
__global__ __launch_bounds__(256) void cvt_t_kernel(const float* __restrict__ in,
                                                    __bf16* __restrict__ out,
                                                    int Kd, int Nd) {
    int i = blockIdx.x * blockDim.x + threadIdx.x;
    int total = Kd * Nd;
    int stride = gridDim.x * blockDim.x;
    for (; i < total; i += stride) {
        int k = i % Kd;
        int n = i / Kd;
        out[i] = (__bf16)in[(size_t)k * Nd + n];   // out[n*Kd + k]
    }
}

// ---------------------------------------------------------------- mask @ V term
// reference adds mask post-softmax: ctx += sum_k mask[p,k] * V[p,h,k,:]
// V stored transposed: Vt[(ph*HS + hs)*NN + tok]  -> contiguous per thread
__global__ __launch_bounds__(64) void maskv_kernel(const float* __restrict__ mask,
                                                   const __bf16* __restrict__ Vt,
                                                   float* __restrict__ maskV) {
    int ph = blockIdx.x;
    int p  = ph / HH;
    int s  = threadIdx.x;
    const float*  mp = mask + (size_t)p * NN;
    const __bf16* vp = Vt + ((size_t)ph * HS + s) * NN;
    float acc = 0.f;
    for (int k = 0; k < NN; ++k) acc += mp[k] * (float)vp[k];
    maskV[ph * HS + s] = acc;
}

// ---------------------------------------------------------------- WMMA GEMM
// C(M x Nd) = A(M x 768) @ BT(Nd x 768)^T ; both tiles [128][32], async double-buffered
// MODE 0: QKV epilogue (bias, fold SCALE into Q, scatter Q/K row-major + V transposed)
// MODE 1: proj epilogue (bias, f32 out)
template <int MODE>
__global__ __launch_bounds__(256) void gemm_bf16_kernel(
    const __bf16* __restrict__ A, const __bf16* __restrict__ BT,
    const float* __restrict__ bias,
    __bf16* __restrict__ Qo, __bf16* __restrict__ Ko, __bf16* __restrict__ Vo,
    float* __restrict__ out, int Ndim) {
    const int Kdim = DD;
    __shared__ __bf16 As[2][128][40];   // stride 80B, 16B aligned segments
    __shared__ __bf16 Bs[2][128][40];

    int tid  = threadIdx.x;
    int wid  = tid >> 5, lane = tid & 31, half = lane >> 4, l16 = lane & 15;
    int waveM = wid >> 1, waveN = wid & 1;
    int m0 = blockIdx.y * 128, n0 = blockIdx.x * 128;

    v8f c[2][4];
    #pragma unroll
    for (int s = 0; s < 2; ++s)
        #pragma unroll
        for (int n = 0; n < 4; ++n) c[s][n] = (v8f){};

    int arow = tid >> 1, aseg = tid & 1;   // 128 rows x 2 segs of 16 elems
    const __bf16* Ap = A  + (size_t)(m0 + arow) * Kdim + aseg * 16;
    const __bf16* Bp = BT + (size_t)(n0 + arow) * Kdim + aseg * 16;

    // prefetch k-tile 0 into buffer 0
    async_b128(&As[0][arow][aseg * 16],     Ap);
    async_b128(&As[0][arow][aseg * 16 + 8], Ap + 8);
    async_b128(&Bs[0][arow][aseg * 16],     Bp);
    async_b128(&Bs[0][arow][aseg * 16 + 8], Bp + 8);

    const int T = Kdim / 32;
    for (int kt = 0; kt < T; ++kt) {
        async_wait0();
        __syncthreads();                       // tile kt resident for all waves
        int cb = kt & 1;
        if (kt + 1 < T) {                      // prefetch kt+1 into other buffer
            int nb = cb ^ 1;
            const __bf16* Ap2 = Ap + (kt + 1) * 32;
            const __bf16* Bp2 = Bp + (kt + 1) * 32;
            async_b128(&As[nb][arow][aseg * 16],     Ap2);
            async_b128(&As[nb][arow][aseg * 16 + 8], Ap2 + 8);
            async_b128(&Bs[nb][arow][aseg * 16],     Bp2);
            async_b128(&Bs[nb][arow][aseg * 16 + 8], Bp2 + 8);
        }

        v16bf aF[2];
        #pragma unroll
        for (int s = 0; s < 2; ++s) {
            int row = waveM * 32 + s * 16 + l16;
            uint4* au = (uint4*)&aF[s];
            au[0] = *(const uint4*)&As[cb][row][half * 8];
            au[1] = *(const uint4*)&As[cb][row][16 + half * 8];
        }
        #pragma unroll
        for (int n = 0; n < 4; ++n) {
            int col = waveN * 64 + n * 16 + l16;
            v16bf bF;
            uint4* bu = (uint4*)&bF;
            bu[0] = *(const uint4*)&Bs[cb][col][half * 8];
            bu[1] = *(const uint4*)&Bs[cb][col][16 + half * 8];
            #pragma unroll
            for (int s = 0; s < 2; ++s)
                c[s][n] = __builtin_amdgcn_wmma_f32_16x16x32_bf16(
                    false, aF[s], false, bF, (short)0, c[s][n], false, false);
        }
    }

    #pragma unroll
    for (int s = 0; s < 2; ++s)
        #pragma unroll
        for (int n = 0; n < 4; ++n)
            #pragma unroll
            for (int r = 0; r < 8; ++r) {
                int gm = m0 + waveM * 32 + s * 16 + half * 8 + r;
                int gn = n0 + waveN * 64 + n * 16 + l16;
                float v = c[s][n][r] + bias[gn];
                if (MODE == 0) {
                    int w = gn / DD, rem = gn - w * DD;
                    int h = rem >> 6, hs = rem & 63;
                    int p = gm >> 10, tok = gm & 1023;
                    size_t ph = (size_t)(p * HH + h);
                    if (w == 0)      Qo[(ph * NN + tok) * HS + hs] = (__bf16)(v * SCALE);
                    else if (w == 1) Ko[(ph * NN + tok) * HS + hs] = (__bf16)v;
                    else             Vo[(ph * HS + hs) * NN + tok] = (__bf16)v;  // transposed
                } else {
                    out[(size_t)gm * Ndim + gn] = v;
                }
            }
}

// ---------------------------------------------------------------- flash attention
// block = 8 waves; wave w owns 32 query rows; keys in tiles of 32 (async-staged)
__global__ __launch_bounds__(256) void attn_kernel(const __bf16* __restrict__ Q,
                                                   const __bf16* __restrict__ Kt,
                                                   const __bf16* __restrict__ Vt,
                                                   const float* __restrict__ maskV,
                                                   __bf16* __restrict__ ctx) {
    __shared__ __bf16 Klds[32][72];       // [key][hs]   stride 144B
    __shared__ __bf16 Vts[64][40];        // [hs][key]   stride 80B
    __shared__ __bf16 Plds[8][32][40];    // per-wave P  [row][key]

    int bid = blockIdx.x;                 // (p*H+h)*4 + qb
    int qb = bid & 3, ph = bid >> 2;
    int pi = ph / HH, h = ph - pi * HH;
    int tid = threadIdx.x, wid = tid >> 5, lane = tid & 31;
    int half = lane >> 4, l16 = lane & 15;

    const __bf16* Qb  = Q  + (size_t)ph * NN * HS;
    const __bf16* Kb  = Kt + (size_t)ph * NN * HS;
    const __bf16* Vtb = Vt + (size_t)ph * HS * NN;
    int qBase = qb * 256 + wid * 32;

    // Q fragments resident in registers (scale pre-folded into Q)
    v16bf qa[2][2];
    #pragma unroll
    for (int mf = 0; mf < 2; ++mf)
        #pragma unroll
        for (int kbi = 0; kbi < 2; ++kbi) {
            int row = qBase + mf * 16 + l16;
            const __bf16* qp = Qb + (size_t)row * HS + kbi * 32;
            uint4* au = (uint4*)&qa[mf][kbi];
            au[0] = *(const uint4*)(qp + half * 8);
            au[1] = *(const uint4*)(qp + 16 + half * 8);
        }

    float mrun[2][8], lrun[2][8];
    v8f o[2][4];
    #pragma unroll
    for (int mf = 0; mf < 2; ++mf) {
        #pragma unroll
        for (int r = 0; r < 8; ++r) { mrun[mf][r] = -1e30f; lrun[mf][r] = 0.f; }
        #pragma unroll
        for (int n = 0; n < 4; ++n) o[mf][n] = (v8f){};
    }

    int krow = tid >> 3, kseg = tid & 7;  // K tile: 32 rows x 8 segs of 8
    int vrow = tid >> 2, vseg = tid & 3;  // V tile: 64 rows x 4 segs of 8

    for (int kt0 = 0; kt0 < NN; kt0 += 32) {
        __syncthreads();                  // all reads of K/V/P from prev iter done
        async_b128(&Klds[krow][kseg * 8], Kb + (size_t)(kt0 + krow) * HS + kseg * 8);
        async_b128(&Vts[vrow][vseg * 8],  Vtb + (size_t)vrow * NN + kt0 + vseg * 8);
        async_wait0();
        __syncthreads();

        // S = Q @ K^T (32 rows x 32 keys per wave)
        v8f sc[2][2];
        #pragma unroll
        for (int mf = 0; mf < 2; ++mf)
            #pragma unroll
            for (int sub = 0; sub < 2; ++sub) {
                v8f acc = (v8f){};
                #pragma unroll
                for (int kbi = 0; kbi < 2; ++kbi) {
                    int key = sub * 16 + l16;
                    v16bf bF;
                    uint4* bu = (uint4*)&bF;
                    bu[0] = *(const uint4*)&Klds[key][kbi * 32 + half * 8];
                    bu[1] = *(const uint4*)&Klds[key][kbi * 32 + 16 + half * 8];
                    acc = __builtin_amdgcn_wmma_f32_16x16x32_bf16(
                        false, qa[mf][kbi], false, bF, (short)0, acc, false, false);
                }
                sc[mf][sub] = acc;
            }

        // online softmax (C layout: row = half*8+r, col = l16)
        #pragma unroll
        for (int mf = 0; mf < 2; ++mf) {
            float alpha[8];
            #pragma unroll
            for (int r = 0; r < 8; ++r) {
                float tm = fmaxf(sc[mf][0][r], sc[mf][1][r]);
                #pragma unroll
                for (int msk = 1; msk < 16; msk <<= 1)
                    tm = fmaxf(tm, __shfl_xor(tm, msk, 32));
                float mnew = fmaxf(mrun[mf][r], tm);
                float p0 = __expf(sc[mf][0][r] - mnew);
                float p1 = __expf(sc[mf][1][r] - mnew);
                float ts = p0 + p1;
                #pragma unroll
                for (int msk = 1; msk < 16; msk <<= 1)
                    ts += __shfl_xor(ts, msk, 32);
                alpha[r] = __expf(mrun[mf][r] - mnew);
                lrun[mf][r] = lrun[mf][r] * alpha[r] + ts;
                mrun[mf][r] = mnew;
                sc[mf][0][r] = p0;
                sc[mf][1][r] = p1;
            }
            #pragma unroll
            for (int n = 0; n < 4; ++n)
                #pragma unroll
                for (int r = 0; r < 8; ++r) o[mf][n][r] *= alpha[r];
            #pragma unroll
            for (int r = 0; r < 8; ++r) {
                int row = mf * 16 + half * 8 + r;
                Plds[wid][row][l16]      = (__bf16)sc[mf][0][r];
                Plds[wid][row][16 + l16] = (__bf16)sc[mf][1][r];
            }
        }
        __syncthreads();

        // O += P @ V
        #pragma unroll
        for (int mf = 0; mf < 2; ++mf) {
            int row = mf * 16 + l16;
            v16bf pa;
            uint4* au = (uint4*)&pa;
            au[0] = *(const uint4*)&Plds[wid][row][half * 8];
            au[1] = *(const uint4*)&Plds[wid][row][16 + half * 8];
            #pragma unroll
            for (int n = 0; n < 4; ++n) {
                int col = n * 16 + l16;
                v16bf bF;
                uint4* bu = (uint4*)&bF;
                bu[0] = *(const uint4*)&Vts[col][half * 8];
                bu[1] = *(const uint4*)&Vts[col][16 + half * 8];
                o[mf][n] = __builtin_amdgcn_wmma_f32_16x16x32_bf16(
                    false, pa, false, bF, (short)0, o[mf][n], false, false);
            }
        }
    }

    // normalize, add post-softmax-mask term, write ctx (bf16, [p*N+tok][h*64+hs])
    #pragma unroll
    for (int mf = 0; mf < 2; ++mf)
        #pragma unroll
        for (int n = 0; n < 4; ++n)
            #pragma unroll
            for (int r = 0; r < 8; ++r) {
                float val = o[mf][n][r] / lrun[mf][r] + maskV[ph * HS + n * 16 + l16];
                int row = qBase + mf * 16 + half * 8 + r;
                size_t oi = ((size_t)(pi * NN + row)) * DD + h * HS + n * 16 + l16;
                ctx[oi] = (__bf16)val;
            }
}

// ---------------------------------------------------------------- launch
extern "C" void kernel_launch(void* const* d_in, const int* in_sizes, int n_in,
                              void* d_out, int out_size, void* d_ws, size_t ws_size,
                              hipStream_t stream) {
    const float* pixel = (const float*)d_in[0];
    const float* amask = (const float*)d_in[1];
    const float* qkvw  = (const float*)d_in[2];
    const float* qkvb  = (const float*)d_in[3];
    const float* projw = (const float*)d_in[4];
    const float* projb = (const float*)d_in[5];
    float* out = (float*)d_out;

    char* ws = (char*)d_ws;
    const size_t M = (size_t)PP * NN;           // 8192
    size_t o = 0;
    __bf16* Xb  = (__bf16*)(ws + o); o += M * DD * 2;
    __bf16* WqT = (__bf16*)(ws + o); o += (size_t)DD * 3 * DD * 2;   // [2304][768]
    __bf16* WpT = (__bf16*)(ws + o); o += (size_t)DD * DD * 2;       // [768][768]
    __bf16* Qb  = (__bf16*)(ws + o); o += M * DD * 2;                // [ph][tok][hs]
    __bf16* Kb  = (__bf16*)(ws + o); o += M * DD * 2;                // [ph][tok][hs]
    __bf16* Vt  = (__bf16*)(ws + o); o += M * DD * 2;                // [ph][hs][tok]
    __bf16* Cb  = (__bf16*)(ws + o); o += M * DD * 2;                // [tok][d]
    float*  mV  = (float*)(ws + o);  o += (size_t)PP * HH * HS * 4;

    cvt_kernel<<<1024, 256, 0, stream>>>(pixel, Xb, (int)(M * DD));
    cvt_t_kernel<<<1024, 256, 0, stream>>>(qkvw, WqT, DD, 3 * DD);
    cvt_t_kernel<<<1024, 256, 0, stream>>>(projw, WpT, DD, DD);

    gemm_bf16_kernel<0><<<dim3(3 * DD / 128, M / 128), 256, 0, stream>>>(
        Xb, WqT, qkvb, Qb, Kb, Vt, nullptr, 3 * DD);

    maskv_kernel<<<PP * HH, 64, 0, stream>>>(amask, Vt, mV);

    attn_kernel<<<PP * HH * (NN / 256), 256, 0, stream>>>(Qb, Kb, Vt, mV, Cb);

    gemm_bf16_kernel<1><<<dim3(DD / 128, M / 128), 256, 0, stream>>>(
        Cb, WpT, projb, nullptr, nullptr, nullptr, out, DD);
}
